// swin_transformer_angular_tan_47090021433562
// MI455X (gfx1250) — matrix operand order repaired and checked by reading, not compiled, not run on hardware
//
#include <hip/hip_runtime.h>

// ---------------------------------------------------------------------------
// CDNA5 (gfx1250) wave32 WMMA implementation of windowed angular attention.
// Uses v_wmma_f32_16x16x32_bf16 + global_load_async_to_lds_b128 (ASYNCcnt).
// ---------------------------------------------------------------------------

typedef __bf16 v16bf __attribute__((ext_vector_type(16)));
typedef float v8f  __attribute__((ext_vector_type(8)));

union Frag16 {
  v16bf v;
  unsigned int u[8];
  unsigned short h[16];
};

__device__ __forceinline__ unsigned short f32_to_bf16(float f) {
  union { float f; unsigned int u; } c; c.f = f;
  unsigned int u = c.u;
  u += 0x7FFFu + ((u >> 16) & 1u);   // round-to-nearest-even
  return (unsigned short)(u >> 16);
}
__device__ __forceinline__ unsigned int pack_bf16x2(float lo, float hi) {
  return (unsigned int)f32_to_bf16(lo) | ((unsigned int)f32_to_bf16(hi) << 16);
}

// async DMA: global (saddr base + 32-bit voff) -> LDS (per-lane byte addr), 16B
__device__ __forceinline__ void async_ld_b128(unsigned int lds_addr,
                                              unsigned int voff,
                                              unsigned long long base) {
  asm volatile("global_load_async_to_lds_b128 %0, %1, %2"
               :: "v"(lds_addr), "v"(voff), "s"(base) : "memory");
}
__device__ __forceinline__ void wait_async0() {
  asm volatile("s_wait_asynccnt 0x0" ::: "memory");
}

// ------------------------- prep: f32 -> bf16 -------------------------------
__global__ void k_cvt_bf16(const float* __restrict__ in,
                           unsigned short* __restrict__ out, int n) {
  int i = blockIdx.x * blockDim.x + threadIdx.x;
  if (i < n) out[i] = f32_to_bf16(in[i]);
}

// ------------------- prep: combined angular bias table ---------------------
__global__ void k_bias(const float* __restrict__ theta_max,
                       const float* __restrict__ a_p, const float* __restrict__ b_p,
                       const float* __restrict__ a_r, const float* __restrict__ b_r,
                       float* __restrict__ bias) {
  int idx = blockIdx.x * blockDim.x + threadIdx.x;      // 16*64*64
  int h = idx >> 12, ij = idx & 4095;
  int i = ij >> 6, j = ij & 63;
  int yi = i >> 3, xi = i & 7, yj = j >> 3, xj = j & 7;
  int rr = yi - yj, ra = xi - xj;
  int idr = rr % 15; if (idr < 0) idr += 15;            // python %
  int ida = ra % 15; if (ida < 0) ida += 15;
  float ang_a = (float)ra * 0.19634954084936207f;       // 2*pi/RES_W
  float ang_r = (float)rr * (theta_max[0] * 0.03125f);  // theta_max/RES_H
  float v = a_p[ida * 16 + h] * __cosf(ang_a) + b_p[ida * 16 + h] * __sinf(ang_a)
          + a_r[idr * 16 + h] * __cosf(ang_r) + b_r[idr * 16 + h] * __sinf(ang_r);
  bias[idx] = v;
}

// -------- fragment gathers per documented 16-bit WMMA VGPR layouts ---------
__device__ __forceinline__ int a_kofs(int r, int hi) {
  return 2 * r + ((r & 4) ? 8 : 0) + (hi ? 8 : 0);   // A 16x32 MxK
}
__device__ __forceinline__ int b_kofs(int r, int hi) {
  return 2 * r + ((r & 4) ? 8 : 0) + (hi ? 8 : 0);   // B 32x16 KxN
}

#define LSTR 40   // LDS tile row stride (ushorts): 80B = 16B-aligned, 20 banks
                  // -> conflict-free (20*r mod 64 distinct for r=0..15)

// --------------------------- QKV GEMM --------------------------------------
// X f32 [65536,512] (f32->bf16 in regs), W bf16 [1536,512] (async->LDS),
// out qkv bf16 [65536,1536]; q channels (n<512) carry SCALE.
// block tile 128(M)x64(N); 8 waves, 32x32 per wave (4 accums).
__global__ __launch_bounds__(256) void k_qkv_gemm(
    const float* __restrict__ X, const unsigned short* __restrict__ W,
    const float* __restrict__ bias, unsigned short* __restrict__ C) {
  __shared__ alignas(16) unsigned short As[128 * LSTR];
  __shared__ alignas(16) unsigned short Bs[64 * LSTR];
  const int K = 512, NW = 1536;
  int m0 = blockIdx.x * 128, n0 = blockIdx.y * 64;
  int tid = threadIdx.x, lane = tid & 31, w = tid >> 5;
  int ln = lane & 15, hi = lane >> 4;
  int wm = (w & 3) * 32, wn = (w >> 2) * 32;
  int arow = tid >> 1, acol = (tid & 1) * 16;   // A coop: 16 f32 / thread
  int brow = tid >> 2, bcol = (tid & 3) * 8;    // B coop: 8 bf16 / thread

  unsigned long long wbase = (unsigned long long)(W + (size_t)n0 * K);
  unsigned int bvoff0 = (unsigned int)((brow * K + bcol) * 2);
  unsigned int blds = (unsigned int)(unsigned long long)&Bs[brow * LSTR + bcol];

  v8f z = v8f{0, 0, 0, 0, 0, 0, 0, 0};
  v8f acc[2][2] = {{z, z}, {z, z}};

  for (int k0 = 0; k0 < K; k0 += 32) {
    const float4* pa =
        reinterpret_cast<const float4*>(X + (size_t)(m0 + arow) * K + k0 + acol);
    float4 a0 = pa[0], a1 = pa[1], a2 = pa[2], a3 = pa[3];
    __syncthreads();   // all waves done reading previous tiles
    uint4* da = reinterpret_cast<uint4*>(&As[arow * LSTR + acol]);
    da[0] = make_uint4(pack_bf16x2(a0.x, a0.y), pack_bf16x2(a0.z, a0.w),
                       pack_bf16x2(a1.x, a1.y), pack_bf16x2(a1.z, a1.w));
    da[1] = make_uint4(pack_bf16x2(a2.x, a2.y), pack_bf16x2(a2.z, a2.w),
                       pack_bf16x2(a3.x, a3.y), pack_bf16x2(a3.z, a3.w));
    async_ld_b128(blds, bvoff0 + (unsigned)(k0 * 2), wbase);
    wait_async0();
    __syncthreads();

    Frag16 fa0, fa1, fb0, fb1;
#pragma unroll
    for (int r = 0; r < 8; ++r) {
      int ka = a_kofs(r, hi), kb = b_kofs(r, hi);
      fa0.u[r] = *reinterpret_cast<const unsigned int*>(&As[(wm + ln) * LSTR + ka]);
      fa1.u[r] = *reinterpret_cast<const unsigned int*>(&As[(wm + 16 + ln) * LSTR + ka]);
      fb0.u[r] = *reinterpret_cast<const unsigned int*>(&Bs[(wn + ln) * LSTR + kb]);
      fb1.u[r] = *reinterpret_cast<const unsigned int*>(&Bs[(wn + 16 + ln) * LSTR + kb]);
    }
    acc[0][0] = __builtin_amdgcn_wmma_f32_16x16x32_bf16(false, fa0.v, false, fb0.v,
                                                        (short)0, acc[0][0], false, false);
    acc[0][1] = __builtin_amdgcn_wmma_f32_16x16x32_bf16(false, fa0.v, false, fb1.v,
                                                        (short)0, acc[0][1], false, false);
    acc[1][0] = __builtin_amdgcn_wmma_f32_16x16x32_bf16(false, fa1.v, false, fb0.v,
                                                        (short)0, acc[1][0], false, false);
    acc[1][1] = __builtin_amdgcn_wmma_f32_16x16x32_bf16(false, fa1.v, false, fb1.v,
                                                        (short)0, acc[1][1], false, false);
  }
#pragma unroll
  for (int i = 0; i < 2; ++i)
#pragma unroll
    for (int j = 0; j < 2; ++j) {
      int n_g = n0 + wn + j * 16 + ln;
      float bv = bias[n_g];
      float scale = (n_g < 512) ? 0.17677669529663687f : 1.0f;  // SCALE on q
#pragma unroll
      for (int r = 0; r < 8; ++r) {
        int m_g = m0 + wm + i * 16 + r + 8 * hi;
        C[(size_t)m_g * NW + n_g] = f32_to_bf16((acc[i][j][r] + bv) * scale);
      }
    }
}

// --------------------------- attention -------------------------------------
__global__ __launch_bounds__(128) void k_attn(
    const unsigned short* __restrict__ QKV,   // bf16 [65536,1536]
    const float* __restrict__ Bias,           // f32 [16][64][64]
    unsigned short* __restrict__ O) {         // bf16 [1024][64][512]
  __shared__ unsigned short Ps[64 * 68];
  int bh = blockIdx.x;
  int b = bh >> 4, h = bh & 15;
  int tid = threadIdx.x, lane = tid & 31, w = tid >> 5;
  int ln = lane & 15, hi = lane >> 4;
  const size_t row0 = (size_t)(b * 64) * 1536;
  const int qo = h * 32, ko = 512 + h * 32, vo = 1024 + h * 32;
  const float* bias = Bias + (size_t)h * 4096;

  Frag16 fq;
#pragma unroll
  for (int r = 0; r < 8; ++r)
    fq.u[r] = *reinterpret_cast<const unsigned int*>(
        &QKV[row0 + (size_t)(w * 16 + ln) * 1536 + qo + a_kofs(r, hi)]);
  v8f zero = v8f{0, 0, 0, 0, 0, 0, 0, 0};
  v8f sc[4];
#pragma unroll
  for (int t = 0; t < 4; ++t) {
    Frag16 fk;
#pragma unroll
    for (int r = 0; r < 8; ++r)
      fk.u[r] = *reinterpret_cast<const unsigned int*>(
          &QKV[row0 + (size_t)(t * 16 + ln) * 1536 + ko + b_kofs(r, hi)]);
    sc[t] = __builtin_amdgcn_wmma_f32_16x16x32_bf16(false, fq.v, false, fk.v,
                                                    (short)0, zero, false, false);
  }
#pragma unroll
  for (int t = 0; t < 4; ++t)
#pragma unroll
    for (int r = 0; r < 8; ++r)
      sc[t][r] += bias[(w * 16 + r + 8 * hi) * 64 + t * 16 + ln];

#pragma unroll
  for (int r = 0; r < 8; ++r) {
    float mx = fmaxf(fmaxf(sc[0][r], sc[1][r]), fmaxf(sc[2][r], sc[3][r]));
#pragma unroll
    for (int m = 1; m < 16; m <<= 1) mx = fmaxf(mx, __shfl_xor(mx, m, 32));
    float sum = 0.f;
#pragma unroll
    for (int t = 0; t < 4; ++t) { sc[t][r] = __expf(sc[t][r] - mx); sum += sc[t][r]; }
#pragma unroll
    for (int m = 1; m < 16; m <<= 1) sum += __shfl_xor(sum, m, 32);
    float inv = 1.0f / sum;
    int i = w * 16 + r + 8 * hi;
#pragma unroll
    for (int t = 0; t < 4; ++t)
      Ps[i * 68 + t * 16 + ln] = f32_to_bf16(sc[t][r] * inv);
  }
  __syncthreads();

  v8f o[2] = {zero, zero};
#pragma unroll
  for (int s = 0; s < 2; ++s) {
    Frag16 fp;
#pragma unroll
    for (int r = 0; r < 8; ++r)
      fp.u[r] = *reinterpret_cast<const unsigned int*>(
          &Ps[(w * 16 + ln) * 68 + a_kofs(r, hi) + 32 * s]);
#pragma unroll
    for (int nt = 0; nt < 2; ++nt) {
      Frag16 fv;
#pragma unroll
      for (int r = 0; r < 8; ++r) {
        int kt = b_kofs(r, hi) + 32 * s;   // token index (K dim)
        int d = nt * 16 + ln;
        fv.h[2 * r]     = QKV[row0 + (size_t)kt * 1536 + vo + d];
        fv.h[2 * r + 1] = QKV[row0 + (size_t)(kt + 1) * 1536 + vo + d];
      }
      o[nt] = __builtin_amdgcn_wmma_f32_16x16x32_bf16(false, fp.v, false, fv.v,
                                                      (short)0, o[nt], false, false);
    }
  }
#pragma unroll
  for (int nt = 0; nt < 2; ++nt)
#pragma unroll
    for (int r = 0; r < 8; ++r) {
      int t = w * 16 + r + 8 * hi;
      int d = nt * 16 + ln;
      O[((size_t)(b * 64 + t)) * 512 + h * 32 + d] = f32_to_bf16(o[nt][r]);
    }
}

// --------------------------- proj GEMM -------------------------------------
// A bf16 [65536,512] (async->LDS), W bf16 [512,512] (async->LDS),
// out f32 [65536,512] (+proj_b). 128x64 tile, 32x32 per wave.
__global__ __launch_bounds__(256) void k_proj_gemm(
    const unsigned short* __restrict__ A, const unsigned short* __restrict__ W,
    const float* __restrict__ bias, float* __restrict__ C) {
  __shared__ alignas(16) unsigned short As[128 * LSTR];
  __shared__ alignas(16) unsigned short Bs[64 * LSTR];
  const int K = 512, NW = 512;
  int m0 = blockIdx.x * 128, n0 = blockIdx.y * 64;
  int tid = threadIdx.x, lane = tid & 31, w = tid >> 5;
  int ln = lane & 15, hi = lane >> 4;
  int wm = (w & 3) * 32, wn = (w >> 2) * 32;
  int arow = tid >> 1, acol = (tid & 1) * 16;   // A coop: 2x b128 / thread
  int brow = tid >> 2, bcol = (tid & 3) * 8;    // B coop: 1x b128 / thread

  unsigned long long abase = (unsigned long long)(A + (size_t)m0 * K);
  unsigned long long wbase = (unsigned long long)(W + (size_t)n0 * K);
  unsigned int avoff0 = (unsigned int)((arow * K + acol) * 2);
  unsigned int bvoff0 = (unsigned int)((brow * K + bcol) * 2);
  unsigned int alds = (unsigned int)(unsigned long long)&As[arow * LSTR + acol];
  unsigned int blds = (unsigned int)(unsigned long long)&Bs[brow * LSTR + bcol];

  v8f z = v8f{0, 0, 0, 0, 0, 0, 0, 0};
  v8f acc[2][2] = {{z, z}, {z, z}};

  for (int k0 = 0; k0 < K; k0 += 32) {
    __syncthreads();   // all waves done reading previous tiles
    unsigned int kb = (unsigned)(k0 * 2);
    async_ld_b128(alds,      avoff0 + kb,      abase);
    async_ld_b128(alds + 16, avoff0 + kb + 16, abase);
    async_ld_b128(blds,      bvoff0 + kb,      wbase);
    wait_async0();
    __syncthreads();

    Frag16 fa0, fa1, fb0, fb1;
#pragma unroll
    for (int r = 0; r < 8; ++r) {
      int ka = a_kofs(r, hi), kkb = b_kofs(r, hi);
      fa0.u[r] = *reinterpret_cast<const unsigned int*>(&As[(wm + ln) * LSTR + ka]);
      fa1.u[r] = *reinterpret_cast<const unsigned int*>(&As[(wm + 16 + ln) * LSTR + ka]);
      fb0.u[r] = *reinterpret_cast<const unsigned int*>(&Bs[(wn + ln) * LSTR + kkb]);
      fb1.u[r] = *reinterpret_cast<const unsigned int*>(&Bs[(wn + 16 + ln) * LSTR + kkb]);
    }
    acc[0][0] = __builtin_amdgcn_wmma_f32_16x16x32_bf16(false, fa0.v, false, fb0.v,
                                                        (short)0, acc[0][0], false, false);
    acc[0][1] = __builtin_amdgcn_wmma_f32_16x16x32_bf16(false, fa0.v, false, fb1.v,
                                                        (short)0, acc[0][1], false, false);
    acc[1][0] = __builtin_amdgcn_wmma_f32_16x16x32_bf16(false, fa1.v, false, fb0.v,
                                                        (short)0, acc[1][0], false, false);
    acc[1][1] = __builtin_amdgcn_wmma_f32_16x16x32_bf16(false, fa1.v, false, fb1.v,
                                                        (short)0, acc[1][1], false, false);
  }
#pragma unroll
  for (int i = 0; i < 2; ++i)
#pragma unroll
    for (int j = 0; j < 2; ++j) {
      int n_g = n0 + wn + j * 16 + ln;
      float bv = bias[n_g];
#pragma unroll
      for (int r = 0; r < 8; ++r) {
        int m_g = m0 + wm + i * 16 + r + 8 * hi;
        C[(size_t)m_g * NW + n_g] = acc[i][j][r] + bv;
      }
    }
}

// ---------------------------------------------------------------------------
extern "C" void kernel_launch(void* const* d_in, const int* in_sizes, int n_in,
                              void* d_out, int out_size, void* d_ws, size_t ws_size,
                              hipStream_t stream) {
  (void)in_sizes; (void)n_in; (void)out_size; (void)ws_size;
  const float* x         = (const float*)d_in[0];
  const float* theta_max = (const float*)d_in[1];
  const float* qkv_w     = (const float*)d_in[2];
  const float* qkv_b     = (const float*)d_in[3];
  const float* proj_w    = (const float*)d_in[4];
  const float* proj_b    = (const float*)d_in[5];
  const float* a_p       = (const float*)d_in[6];
  const float* b_p       = (const float*)d_in[7];
  const float* a_r       = (const float*)d_in[8];
  const float* b_r       = (const float*)d_in[9];
  float* out = (float*)d_out;

  // workspace layout (bytes)
  char* ws = (char*)d_ws;
  unsigned short* qkv1    = (unsigned short*)(ws);              // 65536*1536 bf16
  unsigned short* attn_o  = (unsigned short*)(ws + 201326592);  // 65536*512  bf16
  unsigned short* wqkvb   = (unsigned short*)(ws + 268435456);  // 1536*512   bf16
  unsigned short* wprojb  = (unsigned short*)(ws + 270008320);  // 512*512    bf16
  float*          biasT   = (float*)(ws + 270532608);           // 16*64*64   f32

  k_cvt_bf16<<<3072, 256, 0, stream>>>(qkv_w, wqkvb, 1536 * 512);
  k_cvt_bf16<<<1024, 256, 0, stream>>>(proj_w, wprojb, 512 * 512);
  k_bias<<<256, 256, 0, stream>>>(theta_max, a_p, b_p, a_r, b_r, biasT);
  k_qkv_gemm<<<dim3(512, 24), 256, 0, stream>>>(x, wqkvb, qkv_b, qkv1);
  k_attn<<<16384, 128, 0, stream>>>(qkv1, biasT, attn_o);
  k_proj_gemm<<<dim3(512, 8), 256, 0, stream>>>(attn_o, wprojb, proj_b, out);
}
